// ScaledDotProdAttention_70806830842340
// MI455X (gfx1250) — matrix operand order repaired and checked by reading, not compiled, not run on hardware
//
#include <hip/hip_runtime.h>
#include <hip/hip_bf16.h>

// ---------------------------------------------------------------------------
// Masked scaled-dot-product attention for MI455X (gfx1250, wave32, WMMA).
// Pipeline: (1) zero N*M bitmask in d_ws, (2) scatter mask bits,
// (3) flash-attention with v_wmma_f32_16x16x32_bf16 for both GEMMs.
// Round 2: BN=128->64 + non-resident Q fragments to get VGPR count under 256
// (kill s_set_vgpr_msb mode, allow 2-3 waves/SIMD).
// ---------------------------------------------------------------------------

typedef __attribute__((ext_vector_type(16))) __bf16 v16bf;
typedef __attribute__((ext_vector_type(8)))  float  v8f;

struct __align__(16) F4 { float x, y, z, w; };
struct __align__(16) U4 { unsigned x, y, z, w; };
struct __align__(8)  U2 { unsigned x, y; };

constexpr int Nq = 8192, Mk = 8192, Dh = 128, Dv = 128;
constexpr int BM = 128;           // query rows per block
constexpr int BN = 64;            // key cols per iteration
constexpr int NW = 8;             // waves per block (256 threads, wave32)
constexpr int NT_S = BN / 16;     // 4  S col-tiles per wave
constexpr int NT_O = Dv / 16;     // 8  O col-tiles per wave
constexpr int KC_S = Dh / 32;     // 4  K-chunks for Q.K^T
constexpr int KC_O = BN / 32;     // 2  K-chunks for P.V
constexpr int MASK_WORDS_PER_ROW = Mk / 32;

constexpr size_t SMEM_BYTES =
    (size_t)(BM * Dh + BN * Dh + Dv * BN + NW * 16 * BN) * 2 /*bf16*/ +
    (size_t)(BM * (BN / 32)) * 4 /*mask words*/;   // = 82944 B -> 3 WGs / 320KB WGP

__device__ __forceinline__ unsigned short bf16s(float f) {
    union { float f; unsigned u; } x; x.f = f;
    unsigned r = x.u + 0x7FFFu + ((x.u >> 16) & 1u);      // round-to-nearest-even
    return (unsigned short)(r >> 16);
}
__device__ __forceinline__ unsigned pk_bf16(float a, float b) {
    return (unsigned)bf16s(a) | ((unsigned)bf16s(b) << 16);
}
__device__ __forceinline__ v8f vzero8() {
    v8f v;
#pragma unroll
    for (int i = 0; i < 8; ++i) v[i] = 0.0f;
    return v;
}
// A-fragment (16x32 bf16): elems 0..7 at base, 8..15 at base+16 (ushort units)
__device__ __forceinline__ v16bf ld_frag_a(const unsigned short* s, int base) {
    union { v16bf v; U4 q[2]; } u;
    u.q[0] = *(const U4*)(s + base);
    u.q[1] = *(const U4*)(s + base + 16);
    return u.v;
}
// B-fragment (32x16 bf16): elems 0..7 at base, 8..15 at base+16 (ushort units)
__device__ __forceinline__ v16bf ld_frag_b(const unsigned short* s, int base) {
    union { v16bf v; U4 q[2]; } u;
    u.q[0] = *(const U4*)(s + base);
    u.q[1] = *(const U4*)(s + base + 16);
    return u.v;
}

// ------------------------------- kernels -----------------------------------

__global__ void zero_u32_kernel(unsigned* __restrict__ p, int n) {
    for (int i = blockIdx.x * blockDim.x + threadIdx.x; i < n;
         i += gridDim.x * blockDim.x)
        p[i] = 0u;
}

__global__ void scatter_mask_kernel(const int* __restrict__ rows,
                                    const int* __restrict__ cols,
                                    int n, unsigned* __restrict__ bits) {
    int i = blockIdx.x * blockDim.x + threadIdx.x;
    if (i < n) {
        int r = rows[i], c = cols[i];
        atomicOr(&bits[(size_t)r * MASK_WORDS_PER_ROW + (c >> 5)],
                 1u << (c & 31));
    }
}

__global__ __launch_bounds__(NW * 32, 1)
void fattn_kernel(const float* __restrict__ Qg, const float* __restrict__ Kg,
                  const float* __restrict__ Vg,
                  const unsigned* __restrict__ maskBits,
                  float* __restrict__ Og) {
    extern __shared__ unsigned char smem_raw[];
    unsigned short* sQ  = (unsigned short*)smem_raw;            // [BM][Dh] bf16
    unsigned short* sK  = sQ  + BM * Dh;                        // [BN][Dh] bf16
    unsigned short* sVt = sK  + BN * Dh;                        // [Dv][BN] bf16 (transposed)
    unsigned short* sP  = sVt + (size_t)Dv * BN;                // [NW][16][BN] bf16
    unsigned*       sMask = (unsigned*)(sP + (size_t)NW * 16 * BN); // [BM][BN/32]

    const int tid  = threadIdx.x;
    const int lane = tid & 31;
    const int wave = tid >> 5;
    const int half = lane >> 4;       // 0: lanes 0-15, 1: lanes 16-31
    const int mrow = lane & 15;       // A-row / B-col within 16x16 tile
    const int rowBlock = blockIdx.x * BM;

    // ---- Q tile: global f32 -> LDS bf16 (row-major [row][dim]), loaded once
#pragma unroll
    for (int i = tid; i < BM * Dh / 4; i += NW * 32) {
        int qr = i >> 5;                 // Dh/4 = 32 float4 per row
        int dc = (i & 31) << 2;
        F4 f = *(const F4*)&Qg[(size_t)(rowBlock + qr) * Dh + dc];
        U2 p; p.x = pk_bf16(f.x, f.y); p.y = pk_bf16(f.z, f.w);
        *(U2*)&sQ[qr * Dh + dc] = p;
    }

    v8f accO[NT_O];
#pragma unroll
    for (int t = 0; t < NT_O; ++t) accO[t] = vzero8();
    float mRun[8], lRun[8];
#pragma unroll
    for (int r = 0; r < 8; ++r) { mRun[r] = -INFINITY; lRun[r] = 0.0f; }

    const float SCALE = 0.08838834764831845f;   // 1/sqrt(128)
    const float LOG2E = 1.4426950408889634f;

    __syncthreads();

    for (int j = 0; j < Mk; j += BN) {
        // prefetch next K/V block into GL2 while we work on this one
        if (j + BN < Mk) {
            __builtin_prefetch(&Kg[(size_t)(j + BN) * Dh], 0, 3);
            __builtin_prefetch(&Vg[(size_t)(j + BN) * Dv], 0, 3);
        }

        // ---- cooperative loads: K (row-major bf16), V (transposed bf16), mask
#pragma unroll
        for (int i = tid; i < BN * Dh / 4; i += NW * 32) {
            int key = i >> 5;
            int dc  = (i & 31) << 2;
            F4 f = *(const F4*)&Kg[(size_t)(j + key) * Dh + dc];
            U2 p; p.x = pk_bf16(f.x, f.y); p.y = pk_bf16(f.z, f.w);
            *(U2*)&sK[key * Dh + dc] = p;
        }
#pragma unroll
        for (int i = tid; i < BN * Dv / 4; i += NW * 32) {
            int key = i >> 5;                // key index within block (0..63)
            int dc  = (i & 31) << 2;         // dv dim start
            F4 f = *(const F4*)&Vg[(size_t)(j + key) * Dv + dc];
            sVt[(dc + 0) * BN + key] = bf16s(f.x);
            sVt[(dc + 1) * BN + key] = bf16s(f.y);
            sVt[(dc + 2) * BN + key] = bf16s(f.z);
            sVt[(dc + 3) * BN + key] = bf16s(f.w);
        }
        if (tid < BM * (BN / 32)) {          // 256 mask words
            int rl = tid >> 1, wc = tid & 1;
            sMask[tid] = maskBits[(size_t)(rowBlock + rl) * MASK_WORDS_PER_ROW +
                                  (j >> 5) + wc];
        }
        __syncthreads();

        // ---- S = (Q . K^T) * scale, masked: NT_S col-tiles x KC_S K-chunks
        v8f sAcc[NT_S];
#pragma unroll
        for (int t = 0; t < NT_S; ++t) {
            v8f c = vzero8();
#pragma unroll
            for (int c4 = 0; c4 < KC_S; ++c4) {
                v16bf a = ld_frag_a(sQ, (wave * 16 + mrow) * Dh + c4 * 32 + half * 8);
                v16bf b = ld_frag_b(sK, (t * 16 + mrow) * Dh + c4 * 32 + half * 8);
                c = __builtin_amdgcn_wmma_f32_16x16x32_bf16(
                        false, a, false, b, (short)0, c, false, false);
            }
            sAcc[t] = c;
        }

        // scale + scatter-mask (-inf) using LDS bitmask tile
#pragma unroll
        for (int t = 0; t < NT_S; ++t) {
            int cl = t * 16 + mrow;
            unsigned bit = 1u << (cl & 31);
#pragma unroll
            for (int r = 0; r < 8; ++r) {
                int rl = wave * 16 + r + 8 * half;
                float s = sAcc[t][r] * SCALE;
                if (sMask[rl * (BN / 32) + (cl >> 5)] & bit) s = -INFINITY;
                sAcc[t][r] = s;
            }
        }

        // ---- online softmax: row max/sum via intra-half shuffles
        float mNew[8];
#pragma unroll
        for (int r = 0; r < 8; ++r) {
            float mx = mRun[r];
#pragma unroll
            for (int t = 0; t < NT_S; ++t) mx = fmaxf(mx, sAcc[t][r]);
#pragma unroll
            for (int off = 1; off < 16; off <<= 1)
                mx = fmaxf(mx, __shfl_xor(mx, off));
            mNew[r] = mx;
        }
#pragma unroll
        for (int r = 0; r < 8; ++r) {
            float corr = (mNew[r] == -INFINITY)
                             ? 1.0f
                             : exp2f((mRun[r] - mNew[r]) * LOG2E);
            lRun[r] *= corr;
            mRun[r] = mNew[r];
#pragma unroll
            for (int t = 0; t < NT_O; ++t) accO[t][r] *= corr;
        }
        float rowSum[8];
#pragma unroll
        for (int r = 0; r < 8; ++r) rowSum[r] = 0.0f;
#pragma unroll
        for (int t = 0; t < NT_S; ++t) {
            int cl = t * 16 + mrow;
#pragma unroll
            for (int r = 0; r < 8; ++r) {
                float s = sAcc[t][r];
                float p = (mNew[r] == -INFINITY)
                              ? 0.0f
                              : exp2f((s - mNew[r]) * LOG2E);
                rowSum[r] += p;
                // C-layout -> row-major bf16 P scratch (per-wave region)
                sP[((size_t)wave * 16 + r + 8 * half) * BN + cl] = bf16s(p);
            }
        }
#pragma unroll
        for (int r = 0; r < 8; ++r) {
#pragma unroll
            for (int off = 1; off < 16; off <<= 1)
                rowSum[r] += __shfl_xor(rowSum[r], off);
            lRun[r] += rowSum[r];
        }

        // make this wave's P scratch stores visible to its own ds loads
        asm volatile("s_wait_dscnt 0" ::: "memory");

        // ---- O += P . V  (A = P from scratch, B = V^T contiguous)
        v16bf aP[KC_O];
#pragma unroll
        for (int c4 = 0; c4 < KC_O; ++c4)
            aP[c4] = ld_frag_a(sP, (wave * 16 + mrow) * BN + c4 * 32 + half * 8);
#pragma unroll
        for (int t = 0; t < NT_O; ++t) {
#pragma unroll
            for (int c4 = 0; c4 < KC_O; ++c4) {
                v16bf b = ld_frag_b(sVt, (t * 16 + mrow) * BN + c4 * 32 + half * 8);
                accO[t] = __builtin_amdgcn_wmma_f32_16x16x32_bf16(
                              false, aP[c4], false, b, (short)0, accO[t],
                              false, false);
            }
        }
        __syncthreads();   // before next iteration overwrites sK/sVt/sMask
    }

    // ---- epilogue: normalize and store fp32 output
#pragma unroll
    for (int r = 0; r < 8; ++r) {
        int row = rowBlock + wave * 16 + r + 8 * half;
        float inv = 1.0f / lRun[r];
#pragma unroll
        for (int t = 0; t < NT_O; ++t) {
            int col = t * 16 + mrow;
            Og[(size_t)row * Dv + col] = accO[t][r] * inv;
        }
    }
}

// ------------------------------- launcher ----------------------------------

extern "C" void kernel_launch(void* const* d_in, const int* in_sizes, int n_in,
                              void* d_out, int out_size, void* d_ws, size_t ws_size,
                              hipStream_t stream) {
    const float* Q = (const float*)d_in[0];
    const float* K = (const float*)d_in[1];
    const float* V = (const float*)d_in[2];
    const int* mrows = (const int*)d_in[3];
    const int* mcols = (const int*)d_in[4];
    float* out = (float*)d_out;

    unsigned* maskBits = (unsigned*)d_ws;                 // 8 MB bitmask
    const int maskWords = Nq * MASK_WORDS_PER_ROW;        // 2,097,152 words
    const int nMask = in_sizes[3];

    zero_u32_kernel<<<2048, 256, 0, stream>>>(maskBits, maskWords);
    scatter_mask_kernel<<<(nMask + 255) / 256, 256, 0, stream>>>(
        mrows, mcols, nMask, maskBits);
    fattn_kernel<<<Nq / BM, NW * 32, SMEM_BYTES, stream>>>(
        Q, K, V, maskBits, out);
}
